// MODEL_MNIST_42528766165355
// MI455X (gfx1250) — matrix operand order repaired and checked by reading, not compiled, not run on hardware
//
#include <hip/hip_runtime.h>
#include <hip/hip_bf16.h>
#include <math.h>

typedef __attribute__((ext_vector_type(16))) _Float16 v16h;
typedef __attribute__((ext_vector_type(8)))  float    v8f;
typedef __attribute__((ext_vector_type(4)))  unsigned u32x4;
typedef __attribute__((ext_vector_type(8)))  unsigned u32x8;

#define BATCH 16384
#define CODES 512
#define DIM   64

// ---------------------------------------------------------------------------
// TDM: stage a 32(K) x 16(N) f32 tile of row-major W[K,N] into LDS.
// 2D descriptor (groups 2/3 omitted). tensor dims = remaining extent from the
// tile origin, so the TDM zero-fills K/N remainders (OOB reads return zero).
// LDS receives tile row-major: lds[kk*16 + n].
// ---------------------------------------------------------------------------
__device__ __forceinline__ void tdm_stage_tile(const float* gsrc,
                                               unsigned ldsByteOff,
                                               int rowsRem, int colsRem, int N)
{
    unsigned long long ga = (unsigned long long)(const void*)gsrc;
    u32x4 g0;
    g0[0] = 1u;                                          // count=1, user D#
    g0[1] = ldsByteOff;                                  // lds_addr (bytes)
    g0[2] = (unsigned)ga;                                // global_addr[31:0]
    g0[3] = (unsigned)((ga >> 32) & 0x1FFFFFFu)          // global_addr[56:32]
          | (2u << 30);                                  // type = 2 ("image")
    u32x8 g1;
    g1[0] = (2u << 16);                                  // data_size = 4B
    g1[1] = ((unsigned)colsRem & 0xffffu) << 16;         // tensor_dim0[15:0]
    g1[2] = (((unsigned)colsRem >> 16) & 0xffffu)        // tensor_dim0[31:16]
          | (((unsigned)rowsRem & 0xffffu) << 16);       // tensor_dim1[15:0]
    g1[3] = (((unsigned)rowsRem >> 16) & 0xffffu)        // tensor_dim1[31:16]
          | (16u << 16);                                 // tile_dim0 = 16
    g1[4] = 32u;                                         // tile_dim1=32, tile_dim2=0
    g1[5] = (unsigned)N;                                 // tensor_dim0_stride[31:0]
    g1[6] = 0u;                                          // stride hi / dim1_stride
    g1[7] = 0u;
    asm volatile("tensor_load_to_lds %0, %1" :: "s"(g0), "s"(g1) : "memory");
}

__device__ __forceinline__ unsigned lds_offset(const void* p)
{
    // LDS flat addresses carry the wave-relative byte offset in addr[31:0]
    return (unsigned)(unsigned long long)p;
}

// ---------------------------------------------------------------------------
// Fused GEMM + bias + activation.  C[M,N] = act(A[M,K] @ W[K,N] + b[N])
// f16 WMMA, f32 accumulate. One wave per 16x16 C tile; 8 waves (256 thr)
// share one TDM-staged, double-buffered W tile per k-step.
// grid = ((N+15)/16, M/128).   ACT: 0=id, 1=relu, 2=leaky(0.1), 3=tanh
// ---------------------------------------------------------------------------
template <int ACT>
__global__ __launch_bounds__(256) void gemm_bias_act(
    const float* __restrict__ A, const float* __restrict__ W,
    const float* __restrict__ bias, float* __restrict__ C,
    int M, int N, int K)
{
    __shared__ float ldsB[2][32 * 16];

    const int wave = threadIdx.x >> 5;
    const int lane = threadIdx.x & 31;
    const int mlo  = lane & 15;
    const int hi   = lane >> 4;
    const int n0   = blockIdx.x * 16;
    const int m0   = (blockIdx.y * 8 + wave) * 16;

    const float* Ap   = A + (size_t)(m0 + mlo) * K;
    const int    col  = n0 + mlo;
    const bool   colok = col < N;

    // prologue: stage first W tile (wave 0 issues; TENSORcnt is per-wave)
    if (threadIdx.x < 32) {
        tdm_stage_tile(W + n0, lds_offset(&ldsB[0][0]), K, N - n0, N);
        __builtin_amdgcn_s_wait_tensorcnt(0);
    }
    __syncthreads();

    v8f acc = {};
    for (int k0 = 0; k0 < K; k0 += 32) {
        const int cur = (k0 >> 5) & 1;
        // issue next tile into the other buffer while we compute
        if (threadIdx.x < 32 && (k0 + 32) < K)
            tdm_stage_tile(W + (size_t)(k0 + 32) * N + n0,
                           lds_offset(&ldsB[cur ^ 1][0]),
                           K - (k0 + 32), N - n0, N);

        if ((k0 + 32) < K) __builtin_prefetch(Ap + k0 + 32, 0, 3);

        // A fragment: 16x32 f16 ISA layout
        v16h a;
#pragma unroll
        for (int e = 0; e < 16; ++e) {
            int kk = k0 + ((e < 8) ? 0 : 16) + hi * 8 + (e & 7);
            float v = (kk < K) ? Ap[kk] : 0.0f;
            a[e] = (_Float16)v;
        }
        // B fragment from LDS-staged tile: element e -> K = (e<8?0:16)+hi*8+(e&7), col = mlo
        v16h b;
#pragma unroll
        for (int e = 0; e < 16; ++e)
            b[e] = (_Float16)ldsB[cur][(((e < 8) ? 0 : 16) + hi * 8 + (e & 7)) * 16 + mlo];

        acc = __builtin_amdgcn_wmma_f32_16x16x32_f16(
            false, a, false, b, (short)0, acc, false, false);

        if (threadIdx.x < 32) __builtin_amdgcn_s_wait_tensorcnt(0);
        __syncthreads();
    }

    if (colok) {
        const float bv = bias[col];
#pragma unroll
        for (int r = 0; r < 8; ++r) {
            int row = m0 + r + hi * 8;          // C layout: VGPR r -> M = r + 8*half
            float v = acc[r] + bv;
            if      (ACT == 1) v = v > 0.0f ? v : 0.0f;
            else if (ACT == 2) v = v > 0.0f ? v : 0.1f * v;
            else if (ACT == 3) v = tanhf(v);
            C[(size_t)row * N + col] = v;
        }
    }
}

// ---------------------------------------------------------------------------
// Squared row norms for [rows, 64] f32 matrix.
// ---------------------------------------------------------------------------
__global__ void row_norms_kernel(const float* __restrict__ X,
                                 float* __restrict__ out, int rows)
{
    int r = blockIdx.x * blockDim.x + threadIdx.x;
    if (r >= rows) return;
    const float* p = X + (size_t)r * DIM;
    float s = 0.0f;
#pragma unroll
    for (int i = 0; i < DIM; ++i) { float v = p[i]; s += v * v; }
    out[r] = s;
}

__global__ void init_colmin_kernel(unsigned long long* cm)
{
    int k = blockIdx.x * blockDim.x + threadIdx.x;
    if (k < CODES) cm[k] = ~0ull;
}

__device__ __forceinline__ unsigned ordered_f32(float f)
{
    unsigned u = __float_as_uint(f);
    return (u & 0x80000000u) ? ~u : (u | 0x80000000u);
}

// ---------------------------------------------------------------------------
// VQ nearest-neighbor both ways via WMMA cross-term Z @ E^T.
// One wave per 16-row tile of Z_enc; loops all 32 code tiles.
// row argmin:  min_k ( ||E_k||^2 - 2 Z_b.E_k )  -> idx1[b]
// col argmin:  min_b ( ||Z_b||^2 - 2 Z_b.E_k )  -> atomicMin u64 key per k
// ---------------------------------------------------------------------------
__global__ __launch_bounds__(256) void vq_argmin_kernel(
    const float* __restrict__ Z, const float* __restrict__ E,
    const float* __restrict__ qn, const float* __restrict__ en,
    int* __restrict__ idx1, unsigned long long* __restrict__ colmin)
{
    const int wave = threadIdx.x >> 5;
    const int lane = threadIdx.x & 31;
    const int mlo  = lane & 15;
    const int hi   = lane >> 4;
    const int m0   = (blockIdx.x * 8 + wave) * 16;

    // A fragments for K = 0..31 and 32..63 of Z rows (loaded once, reused 32x)
    const float* Zp = Z + (size_t)(m0 + mlo) * DIM;
    v16h a0, a1;
#pragma unroll
    for (int e = 0; e < 16; ++e) {
        int kk = hi * 16 + e;
        a0[e] = (_Float16)Zp[kk];
        a1[e] = (_Float16)Zp[32 + kk];
    }

    float rowmin[8];
    int   rowidx[8];
#pragma unroll
    for (int r = 0; r < 8; ++r) { rowmin[r] = 3.4e38f; rowidx[r] = 0; }

    for (int ct = 0; ct < CODES / 16; ++ct) {
        const int kcol = ct * 16 + mlo;
        const float* Ep = E + (size_t)kcol * DIM;   // row of E = column of E^T
        v16h b0, b1;
#pragma unroll
        for (int e = 0; e < 16; ++e) {
            b0[e] = (_Float16)Ep[hi * 16 + e];
            b1[e] = (_Float16)Ep[32 + hi * 16 + e];
        }
        v8f acc = {};
        acc = __builtin_amdgcn_wmma_f32_16x16x32_f16(false, a0, false, b0,
                                                     (short)0, acc, false, false);
        acc = __builtin_amdgcn_wmma_f32_16x16x32_f16(false, a1, false, b1,
                                                     (short)0, acc, false, false);

        const float enk = en[kcol];
        float cmin = 3.4e38f; int crow = 0;
#pragma unroll
        for (int r = 0; r < 8; ++r) {
            int   row = m0 + r + hi * 8;
            float dot = acc[r];
            float dr  = enk - 2.0f * dot;           // + ||Z_b||^2 const over k
            if (dr < rowmin[r]) { rowmin[r] = dr; rowidx[r] = kcol; }
            float dc  = qn[row] - 2.0f * dot;       // + ||E_k||^2 const over b
            if (dc < cmin) { cmin = dc; crow = row; }
        }
        // combine the two lane-halves (rows 0-7 vs 8-15 of this column)
        float om = __shfl_xor(cmin, 16, 32);
        int   oi = __shfl_xor(crow, 16, 32);
        if (om < cmin || (om == cmin && oi < crow)) { cmin = om; crow = oi; }
        if (hi == 0) {
            unsigned long long key =
                ((unsigned long long)ordered_f32(cmin) << 32) | (unsigned)crow;
            atomicMin(&colmin[kcol], key);
        }
    }

    // row argmin: reduce across the 16 lanes of each half (xor 1,2,4,8 stay in-half)
#pragma unroll
    for (int r = 0; r < 8; ++r) {
        float m = rowmin[r]; int i = rowidx[r];
#pragma unroll
        for (int d = 1; d < 16; d <<= 1) {
            float om = __shfl_xor(m, d, 32);
            int   oi = __shfl_xor(i, d, 32);
            if (om < m || (om == m && oi < i)) { m = om; i = oi; }
        }
        if (mlo == 0) idx1[m0 + r + hi * 8] = i;
    }
}

__global__ void gather_zdec_kernel(const int* __restrict__ idx1,
                                   const float* __restrict__ E,
                                   float* __restrict__ zdec)
{
    int t = blockIdx.x * blockDim.x + threadIdx.x;   // BATCH*DIM threads
    int b = t >> 6, d = t & 63;
    zdec[t] = E[(size_t)idx1[b] * DIM + d];
}

__global__ void gather_zfe_kernel(const unsigned long long* __restrict__ colmin,
                                  const float* __restrict__ Z,
                                  float* __restrict__ zfe)
{
    int t = blockIdx.x * blockDim.x + threadIdx.x;   // CODES*DIM threads
    int k = t >> 6, d = t & 63;
    int b = (int)(colmin[k] & 0xffffffffull);
    zfe[t] = Z[(size_t)b * DIM + d];
}

// ---------------------------------------------------------------------------
// launch
// ---------------------------------------------------------------------------
extern "C" void kernel_launch(void* const* d_in, const int* in_sizes, int n_in,
                              void* d_out, int out_size, void* d_ws, size_t ws_size,
                              hipStream_t stream)
{
    (void)in_sizes; (void)n_in; (void)out_size; (void)ws_size;

    const float* X  = (const float*)d_in[0];
    const float* W1 = (const float*)d_in[1];  const float* b1 = (const float*)d_in[2];
    const float* W2 = (const float*)d_in[3];  const float* b2 = (const float*)d_in[4];
    const float* W3 = (const float*)d_in[5];  const float* b3 = (const float*)d_in[6];
    const float* W4 = (const float*)d_in[7];  const float* b4 = (const float*)d_in[8];
    const float* E  = (const float*)d_in[9];
    const float* V1 = (const float*)d_in[10]; const float* c1 = (const float*)d_in[11];
    const float* V2 = (const float*)d_in[12]; const float* c2 = (const float*)d_in[13];
    const float* V3 = (const float*)d_in[14]; const float* c3 = (const float*)d_in[15];
    const float* V4 = (const float*)d_in[16]; const float* c4 = (const float*)d_in[17];

    float* out = (float*)d_out;
    // output layout (floats): X_recon | Z_enc | Z_dec | Z_enc_for_embd
    const size_t XR = 0;
    const size_t ZE = (size_t)BATCH * 784;            // 12845056
    const size_t ZD = ZE + (size_t)BATCH * DIM;       // 13893632
    const size_t ZF = ZD + (size_t)BATCH * DIM;       // 14942208
    float* Xrecon = out + XR;
    float* Zenc   = out + ZE;
    float* Zdec   = out + ZD;
    float* Zfe    = out + ZF;

    // workspace layout (floats); ~86 MB
    float* ws = (float*)d_ws;
    float* H1 = ws;                                   // 16384*1000
    float* H3 = ws + (size_t)16384 * 1000;            // 16384*300
    float* qn = H3 + (size_t)16384 * 300;             // 16384
    float* en = qn + 16384;                           // 512
    int*   idx1 = (int*)(en + 512);                   // 16384
    unsigned long long* colmin =
        (unsigned long long*)((char*)(idx1 + 16384)); // 512 (8B aligned)
    // H2 / decoder-g2 alias the X_recon region (dead before X_recon is written)
    float* H2 = out + XR;                             // 16384*500 fits in 16384*784

    dim3 blk(256);
    auto gemm_grid = [](int M, int N) { return dim3((N + 15) / 16, M / 128); };

    // encoder
    gemm_bias_act<1><<<gemm_grid(BATCH, 1000), blk, 0, stream>>>(X,  W1, b1, H1, BATCH, 1000, 784);
    gemm_bias_act<1><<<gemm_grid(BATCH,  500), blk, 0, stream>>>(H1, W2, b2, H2, BATCH,  500, 1000);
    gemm_bias_act<1><<<gemm_grid(BATCH,  300), blk, 0, stream>>>(H2, W3, b3, H3, BATCH,  300, 500);
    gemm_bias_act<0><<<gemm_grid(BATCH,   64), blk, 0, stream>>>(H3, W4, b4, Zenc, BATCH, 64, 300);

    // vector quantization
    row_norms_kernel<<<(BATCH + 255) / 256, blk, 0, stream>>>(Zenc, qn, BATCH);
    row_norms_kernel<<<(CODES + 255) / 256, blk, 0, stream>>>(E,    en, CODES);
    init_colmin_kernel<<<2, blk, 0, stream>>>(colmin);
    vq_argmin_kernel<<<BATCH / 128, blk, 0, stream>>>(Zenc, E, qn, en, idx1, colmin);
    gather_zdec_kernel<<<(BATCH * DIM) / 256, blk, 0, stream>>>(idx1, E, Zdec);
    gather_zfe_kernel<<<(CODES * DIM) / 256, blk, 0, stream>>>(colmin, Zenc, Zfe);

    // decoder (reuse H3 -> g1, H2 -> g2, H1 -> g3)
    gemm_bias_act<2><<<gemm_grid(BATCH,  300), blk, 0, stream>>>(Zdec, V1, c1, H3, BATCH, 300, 64);
    gemm_bias_act<2><<<gemm_grid(BATCH,  500), blk, 0, stream>>>(H3,  V2, c2, H2, BATCH, 500, 300);
    gemm_bias_act<2><<<gemm_grid(BATCH, 1000), blk, 0, stream>>>(H2,  V3, c3, H1, BATCH, 1000, 500);
    gemm_bias_act<3><<<gemm_grid(BATCH,  784), blk, 0, stream>>>(H1,  V4, c4, Xrecon, BATCH, 784, 1000);
}
